// FFSTLSTM_85358180041349
// MI455X (gfx1250) — matrix operand order repaired and checked by reading, not compiled, not run on hardware
//
#include <hip/hip_runtime.h>

// ---------------------------------------------------------------------------
// FFSTLSTM for MI455X (gfx1250, wave32).
// Every conv is implicit GEMM on v_wmma_f32_16x16x32_bf16 (fp32 accumulate).
// Conv inputs are pre-padded bf16 tensors -> the A-fragment gather is fully
// branch-free (no bounds checks, no k-tail guard, no cvt in the hot loop),
// and uses unsigned 32-bit byte offsets so loads take the saddr+voffset form
// (1 VALU per gathered element). One wave computes a 16(W) x NB*16(Cout)
// tile; NB is a template parameter so one A gather feeds up to 4 WMMAs.
// States (h/c/m) are carried in fp32.
// ---------------------------------------------------------------------------

typedef __bf16 bf16;
typedef __attribute__((ext_vector_type(16))) __bf16 v16bf;
typedef __attribute__((ext_vector_type(8)))  float  v8f;

#define KTAB_MAX 8576   // max KB*32 (conv1: 268*32)

__device__ __forceinline__ float sigmoidf_(float x) { return 1.0f / (1.0f + __expf(-x)); }

// ---------------------------------------------------------------------------
// Implicit-GEMM conv on padded bf16 input.
//   src  : [nBatch][Ci][Dp][Hp][Wp] bf16, zero-padded ("same" conv)
//   wpack: [KB][NB][32 lanes][16 bf16]  (B fragments, zero-padded K/N)
//   dst  : [nBatch*D*H*W][Cpad] fp32, row=(n,d,h,w), col=cout, optional relu
// LDS rel-table holds BYTE offsets: rel[k] = (((ci*Dp+zd)*Hp+zy)*Wp+zx)*2 ;
// element address = uniform_base + posByte + rel[k]  (padding cancels).
// rel[k>=K] = 0: loads a live element, but its B row is zero-padded.
// ---------------------------------------------------------------------------
template <int NB>
__global__ __launch_bounds__(32)
void conv_gemm_wmma(const bf16* __restrict__ src, const bf16* __restrict__ wpack,
                    float* __restrict__ dst,
                    long srcBatchStride,
                    int Dp, int Hp, int Wp,
                    int D, int H, int W,
                    int kd, int kh, int kw,
                    int K, int KB, int Cpad, int doRelu)
{
    __shared__ unsigned ktab[KTAB_MAX];
    const int lane = threadIdx.x;
    const int khw  = kh * kw;
    const int kvol = kd * khw;
    const int KP   = KB << 5;
    for (int k = lane; k < KP; k += 32) {
        unsigned rel = 0;
        if (k < K) {
            int ci = k / kvol;  int rr = k - ci * kvol;
            int zd = rr / khw;  rr -= zd * khw;
            int zy = rr / kw;   int zx = rr - zy * kw;
            rel = (unsigned)(((ci * Dp + zd) * Hp + zy) * Wp + zx) << 1;  // bytes
        }
        ktab[k] = rel;
    }
    __syncthreads();

    const int wt = W >> 4;
    int tile = blockIdx.x;
    const int tilesPerN = D * H * wt;
    const int n = tile / tilesPerN;  tile -= n * tilesPerN;
    const int d = tile / (H * wt);   tile -= d * (H * wt);
    const int h = tile / wt;
    const int w0 = (tile - h * wt) << 4;

    const char* sb = (const char*)(src + (long)n * srcBatchStride);  // uniform base
    const int w    = w0 + (lane & 15);             // A row (M) = W position
    const unsigned posByte = ((unsigned)(d * Hp + h) * (unsigned)Wp + (unsigned)w) << 1;
    const int koff = (lane & 16) ? 8 : 0;          // A lane K-offset

    v8f acc[NB];
#pragma unroll
    for (int nb = 0; nb < NB; ++nb) acc[nb] = (v8f){};

    for (int kb = 0; kb < KB; ++kb) {
        const int kbase = (kb << 5) + koff;
        v16bf a;
#pragma unroll
        for (int i = 0; i < 16; ++i)               // branch-free gather, u32 voffset
            a[i] = *(const bf16*)(sb + (posByte + ktab[kbase + i + (i & 8)]));

        const bf16* bp = wpack + ((long)kb * NB * 32 + lane) * 16;
        if (kb + 1 < KB)
            __builtin_prefetch(bp + (long)NB * 512, 0, 0);
#pragma unroll
        for (int nb = 0; nb < NB; ++nb) {
            const v16bf b = *(const v16bf*)(bp + (long)nb * 512);
            acc[nb] = __builtin_amdgcn_wmma_f32_16x16x32_bf16(false, a, false, b,
                                                              (short)0, acc[nb], false, false);
        }
    }

    const long rowBase = (long)n * ((long)D * H * W) + ((long)(d * H + h)) * W + w0;
    const int  lcol    = lane & 15;
#pragma unroll
    for (int nb = 0; nb < NB; ++nb) {
#pragma unroll
        for (int r = 0; r < 8; ++r) {
            int mm  = r + ((lane & 16) >> 1);      // C layout: lanes>=16 hold M=r+8
            float v = acc[nb][r];
            if (doRelu) v = fmaxf(v, 0.0f);
            dst[(rowBase + mm) * (long)Cpad + (nb << 4) + lcol] = v;
        }
    }
}

// ---------------------------------------------------------------------------
// Weight packing into the WMMA B-fragment layout [KB][NB][lane][16].
// ---------------------------------------------------------------------------
__global__ void pack_w_k(const float* __restrict__ w, bf16* __restrict__ wp,
                         int Co, int K, int KB, int NB)
{
    long idx   = (long)blockIdx.x * blockDim.x + threadIdx.x;
    long total = (long)KB * NB * 512;
    if (idx >= total) return;
    int  i    = (int)(idx & 15);
    int  lane = (int)((idx >> 4) & 31);
    long t    = idx >> 9;
    int  nb   = (int)(t % NB);
    int  kb   = (int)(t / NB);
    int  ncol = (nb << 4) + (lane & 15);
    int  k    = (kb << 5) + ((lane & 16) ? 8 : 0) + i + (i & 8);
    float v = 0.0f;
    if (k < K && ncol < Co) v = w[(long)ncol * K + k];
    wp[idx] = (bf16)v;
}

// ---------------------------------------------------------------------------
// Pad-copy fp32 NCDHW -> zero-padded bf16 (interior only; borders pre-zeroed).
// ---------------------------------------------------------------------------
__global__ void pad_in_k(const float* __restrict__ in, bf16* __restrict__ out,
                         int C, int D, int H, int W,
                         int Dp, int Hp, int Wp, int pd, int ph, int pw, long total)
{
    long idx = (long)blockIdx.x * blockDim.x + threadIdx.x;
    if (idx >= total) return;
    int w = (int)(idx % W); long t = idx / W;
    int h = (int)(t % H);   t /= H;
    int d = (int)(t % D);   t /= D;
    int c = (int)t;
    out[(((long)c * Dp + d + pd) * Hp + h + ph) * Wp + w + pw] = (bf16)in[idx];
}

// ---------------------------------------------------------------------------
// 2x2x2 maxpool; input GEMM layout [D*H*W][Cpad] fp32, output scattered bf16
// (strides point into the next conv's padded tensor, or into seqpad).
// ---------------------------------------------------------------------------
__global__ void maxpool3d_k(const float* __restrict__ in, bf16* __restrict__ out,
                            int Cpad, int D, int H, int W,
                            long base, long sc, long sd, long sh, long total)
{
    long idx = (long)blockIdx.x * blockDim.x + threadIdx.x;
    if (idx >= total) return;
    int OW = W >> 1, OH = H >> 1, OD = D >> 1;
    int ow = (int)(idx % OW);  long t = idx / OW;
    int oh = (int)(t % OH);    t /= OH;
    int od = (int)(t % OD);    t /= OD;
    int c  = (int)t;
    float m = -3.402823466e+38f;
    for (int dz = 0; dz < 2; ++dz)
        for (int dy = 0; dy < 2; ++dy)
            for (int dx = 0; dx < 2; ++dx) {
                int d = od * 2 + dz, h = oh * 2 + dy, w = ow * 2 + dx;
                m = fmaxf(m, in[((long)(d * H + h) * W + w) * Cpad + c]);
            }
    out[base + (long)c * sc + (long)od * sd + (long)oh * sh + ow] = (bf16)m;
}

// ---------------------------------------------------------------------------
// LayerNorm over [C,H,W] per sample. in: GEMM layout [B*256][Cpad] fp32,
// out: [B][C][256] fp32. One block per sample.
// ---------------------------------------------------------------------------
__global__ __launch_bounds__(256)
void layernorm_k(const float* __restrict__ in, const float* __restrict__ g,
                 const float* __restrict__ bb, float* __restrict__ out,
                 int C, int Cpad)
{
    __shared__ float rs[256], rq[256];
    int n = blockIdx.x, hw = threadIdx.x;
    const float* row = in + ((long)n * 256 + hw) * Cpad;
    float s = 0.0f, q = 0.0f;
    for (int c = 0; c < C; ++c) { float v = row[c]; s += v; q += v * v; }
    rs[hw] = s; rq[hw] = q; __syncthreads();
    for (int o = 128; o > 0; o >>= 1) {
        if (hw < o) { rs[hw] += rs[hw + o]; rq[hw] += rq[hw + o]; }
        __syncthreads();
    }
    float cnt  = (float)C * 256.0f;
    float mean = rs[0] / cnt;
    float var  = rq[0] / cnt - mean * mean;
    float inv  = rsqrtf(var + 1e-5f);
    for (int c = 0; c < C; ++c) {
        float v = row[c];
        out[(((long)n * C + c) << 8) + hw] = (v - mean) * inv * g[(c << 8) + hw] + bb[(c << 8) + hw];
    }
}

// ---------------------------------------------------------------------------
// ST-LSTM gates. Writes fp32 states (c, m, mem) and bf16 padded copies of
// m and mem=concat(c,m) for the next conv_m / conv_o.
// ---------------------------------------------------------------------------
__global__ void gates_k(const float* __restrict__ xn, const float* __restrict__ hn,
                        const float* __restrict__ mn, float* __restrict__ c,
                        float* __restrict__ m, float* __restrict__ memf,
                        bf16* __restrict__ mpad, bf16* __restrict__ mempad)
{
    int idx = blockIdx.x * blockDim.x + threadIdx.x;
    if (idx >= 16 * 8 * 256) return;
    int hw = idx & 255, ch = (idx >> 8) & 7, b = idx >> 11;
    long X  = ((long)b * 56 + ch) << 8;
    long Hh = ((long)b * 32 + ch) << 8;
    long M  = ((long)b * 24 + ch) << 8;
    float ix  = xn[X + hw],               fx  = xn[X + (8 << 8) + hw];
    float gx  = xn[X + (16 << 8) + hw],   ipx = xn[X + (24 << 8) + hw];
    float fpx = xn[X + (32 << 8) + hw],   gpx = xn[X + (40 << 8) + hw];
    float ih  = hn[Hh + hw],              fh  = hn[Hh + (8 << 8) + hw];
    float gh  = hn[Hh + (16 << 8) + hw];
    float im  = mn[M + hw],               fm  = mn[M + (8 << 8) + hw];
    float gm  = mn[M + (16 << 8) + hw];

    float it = sigmoidf_(ix + ih);
    float ft = sigmoidf_(fx + fh + 1.0f);        // FB = 1.0
    float gt = tanhf(gx + gh);
    float cn = ft * c[idx] + it * gt;
    float itp = sigmoidf_(ipx + im);
    float ftp = sigmoidf_(fpx + fm + 1.0f);
    float gtp = tanhf(gpx + gm);
    float mnw = ftp * m[idx] + itp * gtp;

    c[idx] = cn; m[idx] = mnw;
    long MB = ((long)b * 16 + ch) << 8;
    memf[MB + hw] = cn;
    memf[MB + (8 << 8) + hw] = mnw;

    int y = (hw >> 4) + 7, xw = (hw & 15) + 7;   // padded (30x30) coords
    mpad[((long)(b * 8 + ch) * 30 + y) * 30 + xw] = (bf16)mnw;
    mempad[((long)(b * 16 + ch) * 30 + y) * 30 + xw] = (bf16)cn;
    mempad[((long)(b * 16 + 8 + ch) * 30 + y) * 30 + xw] = (bf16)mnw;
}

// ---------------------------------------------------------------------------
// h_new = sigmoid(o_x + o_h + occ_n) * tanh(conv1x1(mem, wlast)).
// Writes fp32 h (head / precision) and bf16 padded h (next conv_h / conv_x).
// ---------------------------------------------------------------------------
__global__ void hout_k(const float* __restrict__ xn, const float* __restrict__ hn,
                       const float* __restrict__ on, const float* __restrict__ memf,
                       const float* __restrict__ wlast, float* __restrict__ hnew,
                       bf16* __restrict__ hpad)
{
    int idx = blockIdx.x * blockDim.x + threadIdx.x;
    if (idx >= 16 * 8 * 256) return;
    int hw = idx & 255, ch = (idx >> 8) & 7, b = idx >> 11;
    float ox = xn[(((long)b * 56 + 48 + ch) << 8) + hw];
    float oh = hn[(((long)b * 32 + 24 + ch) << 8) + hw];
    float oo = on[(((long)b * 8 + ch) << 8) + hw];
    float ot = sigmoidf_(ox + oh + oo);
    const float* mb = memf + (((long)b * 16) << 8) + hw;
    float s = 0.0f;
#pragma unroll
    for (int k = 0; k < 16; ++k) s += wlast[ch * 16 + k] * mb[(long)k << 8];
    float hv = ot * tanhf(s);
    hnew[idx] = hv;
    int y = (hw >> 4) + 7, xw = (hw & 15) + 7;
    hpad[((long)(b * 8 + ch) * 30 + y) * 30 + xw] = (bf16)hv;
}

// ---------------------------------------------------------------------------
// BatchNorm2d (batch stats, eps 1e-3) + AvgPool2d(2).
// ---------------------------------------------------------------------------
__global__ __launch_bounds__(256)
void bn_pool_k(const float* __restrict__ o, const float* __restrict__ g,
               const float* __restrict__ bb, float* __restrict__ pooled)
{
    __shared__ float rs[256], rq[256];
    int c = blockIdx.x, t = threadIdx.x;
    int b = t >> 4, part = t & 15;
    const float* base = o + (((long)b * 8 + c) << 8) + (part << 4);
    float s = 0.0f, q = 0.0f;
    for (int j = 0; j < 16; ++j) { float v = base[j]; s += v; q += v * v; }
    rs[t] = s; rq[t] = q; __syncthreads();
    for (int off = 128; off > 0; off >>= 1) {
        if (t < off) { rs[t] += rs[t + off]; rq[t] += rq[t + off]; }
        __syncthreads();
    }
    float mean = rs[0] / 4096.0f;
    float var  = rq[0] / 4096.0f - mean * mean;
    float inv  = rsqrtf(var + 1e-3f);
    float gg = g[c], bv = bb[c];
    for (int u = t; u < 1024; u += 256) {
        int bq = u >> 6, yx = u & 63, y = yx >> 3, x = yx & 7;
        const float* src = o + (((long)bq * 8 + c) << 8);
        float a = 0.0f;
        for (int dy = 0; dy < 2; ++dy)
            for (int dx = 0; dx < 2; ++dx)
                a += (src[(y * 2 + dy) * 16 + (x * 2 + dx)] - mean) * inv * gg + bv;
        pooled[(((long)bq * 8 + c) << 6) + y * 8 + x] = a * 0.25f;
    }
}

// ---------------------------------------------------------------------------
// FC(8192->16) + sigmoid -> out[0..15]; BCE vs target -> out[16].
// ---------------------------------------------------------------------------
__global__ __launch_bounds__(256)
void fc_loss_k(const float* __restrict__ flat, const float* __restrict__ fcw,
               const float* __restrict__ fcb, const float* __restrict__ target,
               float* __restrict__ outp)
{
    __shared__ float red[256];
    __shared__ float preds[16];
    int t = threadIdx.x, j = t >> 4, part = t & 15;
    float s = 0.0f;
    for (int q = part; q < 8192; q += 16) s += fcw[(long)j * 8192 + q] * flat[q];
    red[t] = s; __syncthreads();
    for (int off = 8; off > 0; off >>= 1) {
        if (part < off) red[t] += red[t + off];
        __syncthreads();
    }
    if (part == 0) {
        float p = sigmoidf_(red[t] + fcb[j]);
        outp[j] = p; preds[j] = p;
    }
    __syncthreads();
    if (t == 0) {
        float L = 0.0f;
        for (int k = 0; k < 16; ++k) {
            float p  = preds[k];
            float lp = fmaxf(__logf(p), -100.0f);
            float l1 = fmaxf(__logf(1.0f - p), -100.0f);
            L += target[k] * lp + (1.0f - target[k]) * l1;
        }
        outp[16] = -L / 16.0f;
    }
}

__global__ void zero_f32_k(float* __restrict__ p, long n)
{
    long i = (long)blockIdx.x * blockDim.x + threadIdx.x;
    if (i < n) p[i] = 0.0f;
}
__global__ void zero_bf16_k(bf16* __restrict__ p, long n)
{
    long i = (long)blockIdx.x * blockDim.x + threadIdx.x;
    if (i < n) p[i] = (bf16)0.0f;
}

// ---------------------------------------------------------------------------
extern "C" void kernel_launch(void* const* d_in, const int* in_sizes, int n_in,
                              void* d_out, int out_size, void* d_ws, size_t ws_size,
                              hipStream_t stream)
{
    (void)in_sizes; (void)n_in; (void)out_size; (void)ws_size;
    const float* x      = (const float*)d_in[0];
    const float* target = (const float*)d_in[3];
    const float* w0 = (const float*)d_in[4];
    const float* w1 = (const float*)d_in[5];
    const float* w2 = (const float*)d_in[6];
    const float* wx = (const float*)d_in[7];
    const float* wh = (const float*)d_in[8];
    const float* wm = (const float*)d_in[9];
    const float* wo = (const float*)d_in[10];
    const float* wlast = (const float*)d_in[11];
    const float* gx = (const float*)d_in[12];
    const float* bx = (const float*)d_in[13];
    const float* gh = (const float*)d_in[14];
    const float* bh = (const float*)d_in[15];
    const float* gm = (const float*)d_in[16];
    const float* bm = (const float*)d_in[17];
    const float* go = (const float*)d_in[18];
    const float* bo = (const float*)d_in[19];
    const float* bng = (const float*)d_in[20];
    const float* bnb = (const float*)d_in[21];
    const float* fcw = (const float*)d_in[22];
    const float* fcb = (const float*)d_in[23];
    float* outp = (float*)d_out;

    // ---- workspace carve-up (~90 MB total) --------------------------------
    char* base = (char*)d_ws; size_t off = 0;
    auto alloc = [&](size_t bytes) -> void* {
        void* r = base + off; off = (off + bytes + 255) & ~(size_t)255; return r;
    };
    float* s0   = (float*)alloc((size_t)32 * 128 * 128 * 32 * 4);   // conv0 out (per sample)
    float* s1   = (float*)alloc((size_t)16 * 64 * 64 * 16 * 4);
    float* s2   = (float*)alloc((size_t)8 * 32 * 32 * 16 * 4);
    bf16* xpad  = (bf16*)alloc((size_t)3 * 38 * 134 * 134 * 2);     // padded inputs (bf16)
    bf16* p0pad = (bf16*)alloc((size_t)25 * 22 * 70 * 70 * 2);
    bf16* p1pad = (bf16*)alloc((size_t)10 * 14 * 38 * 38 * 2);
    bf16* seqpad = (bf16*)alloc((size_t)4 * 16 * 8 * 30 * 30 * 2);  // [T][B][8][30][30]
    bf16* hpad  = (bf16*)alloc((size_t)4 * 16 * 8 * 30 * 30 * 2);   // [L][B][8][30][30]
    bf16* mpad  = (bf16*)alloc((size_t)16 * 8 * 30 * 30 * 2);
    bf16* mempad = (bf16*)alloc((size_t)16 * 16 * 30 * 30 * 2);
    bf16* w0p = (bf16*)alloc((size_t)33  * 2 * 512 * 2);
    bf16* w1p = (bf16*)alloc((size_t)268 * 1 * 512 * 2);
    bf16* w2p = (bf16*)alloc((size_t)108 * 1 * 512 * 2);
    bf16* wxp = (bf16*)alloc((size_t)57  * 4 * 512 * 2);
    bf16* whp = (bf16*)alloc((size_t)57  * 2 * 512 * 2);
    bf16* wmp = (bf16*)alloc((size_t)57  * 2 * 512 * 2);
    bf16* wop = (bf16*)alloc((size_t)113 * 1 * 512 * 2);
    float* xcc  = (float*)alloc((size_t)4096 * 64 * 4);
    float* xccn = (float*)alloc((size_t)16 * 56 * 256 * 4);
    float* hcc  = (float*)alloc((size_t)4096 * 32 * 4);
    float* hccn = (float*)alloc((size_t)16 * 32 * 256 * 4);
    float* mcc  = (float*)alloc((size_t)4096 * 32 * 4);
    float* mccn = (float*)alloc((size_t)16 * 24 * 256 * 4);
    float* occ  = (float*)alloc((size_t)4096 * 16 * 4);
    float* occn = (float*)alloc((size_t)16 * 8 * 256 * 4);
    float* memf = (float*)alloc((size_t)16 * 16 * 256 * 4);
    float* hs   = (float*)alloc((size_t)4 * 32768 * 4);
    float* cs   = (float*)alloc((size_t)4 * 32768 * 4);
    float* mbuf = (float*)alloc((size_t)32768 * 4);
    float* pooled = (float*)alloc((size_t)16 * 8 * 64 * 4);

    auto zf = [&](float* p, long n) {
        zero_f32_k<<<dim3((unsigned)((n + 255) / 256)), 256, 0, stream>>>(p, n);
    };
    auto zb = [&](bf16* p, long n) {
        zero_bf16_k<<<dim3((unsigned)((n + 255) / 256)), 256, 0, stream>>>(p, n);
    };
    // zero fp32 states (hs,cs,mbuf contiguous) and all padded bf16 tensors
    zf(hs, 9L * 32768);
    zb(xpad,  3L * 38 * 134 * 134);
    zb(p0pad, 25L * 22 * 70 * 70);
    zb(p1pad, 10L * 14 * 38 * 38);
    zb(seqpad, 4L * 16 * 8 * 900);
    zb(hpad,   4L * 16 * 8 * 900);
    zb(mpad,   16L * 8 * 900);
    zb(mempad, 16L * 16 * 900);

    // ---- pack conv weights to bf16 B-fragment layout ----------------------
    auto pack = [&](const float* w, bf16* wp, int Co, int K, int KB, int NB) {
        long tot = (long)KB * NB * 512;
        pack_w_k<<<dim3((unsigned)((tot + 255) / 256)), 256, 0, stream>>>(w, wp, Co, K, KB, NB);
    };
    pack(w0, w0p, 25, 1029, 33, 2);
    pack(w1, w1p, 10, 8575, 268, 1);
    pack(w2, w2p, 8, 3430, 108, 1);
    pack(wx, wxp, 56, 1800, 57, 4);
    pack(wh, whp, 32, 1800, 57, 2);
    pack(wm, wmp, 24, 1800, 57, 2);
    pack(wo, wop, 8, 3600, 113, 1);

    // ---- conv3d backbone, per sample --------------------------------------
    for (int n = 0; n < 16; ++n) {
        { long tot = 3L * 32 * 128 * 128;
          pad_in_k<<<dim3((unsigned)((tot + 255) / 256)), 256, 0, stream>>>(
              x + (size_t)n * tot, xpad, 3, 32, 128, 128, 38, 134, 134, 3, 3, 3, tot); }
        conv_gemm_wmma<2><<<dim3(32 * 128 * 8), 32, 0, stream>>>(
            xpad, w0p, s0, 0L, 38, 134, 134, 32, 128, 128, 7, 7, 7, 1029, 33, 32, 1);
        { long tot = 25L * 16 * 64 * 64;    // -> p0pad [25][22][70][70] interior
          maxpool3d_k<<<dim3((unsigned)((tot + 255) / 256)), 256, 0, stream>>>(
              s0, p0pad, 32, 32, 128, 128, 3L * 4900 + 3 * 70 + 3, 107800L, 4900L, 70L, tot); }
        conv_gemm_wmma<1><<<dim3(16 * 64 * 4), 32, 0, stream>>>(
            p0pad, w1p, s1, 0L, 22, 70, 70, 16, 64, 64, 7, 7, 7, 8575, 268, 16, 1);
        { long tot = 10L * 8 * 32 * 32;     // -> p1pad [10][14][38][38] interior
          maxpool3d_k<<<dim3((unsigned)((tot + 255) / 256)), 256, 0, stream>>>(
              s1, p1pad, 16, 16, 64, 64, 3L * 1444 + 3 * 38 + 3, 20216L, 1444L, 38L, tot); }
        conv_gemm_wmma<1><<<dim3(8 * 32 * 2), 32, 0, stream>>>(
            p1pad, w2p, s2, 0L, 14, 38, 38, 8, 32, 32, 7, 7, 7, 3430, 108, 16, 1);
        { long tot = 8L * 4 * 16 * 16;      // -> seqpad [T][B][8][30][30] interior (t = od)
          maxpool3d_k<<<dim3((unsigned)((tot + 255) / 256)), 256, 0, stream>>>(
              s2, seqpad, 16, 8, 32, 32, (long)n * 7200 + 7 * 30 + 7, 900L, 115200L, 30L, tot); }
    }

    // ---- ST-LSTM stack: T=4 x NL=4, shared weights ------------------------
    for (int t = 0; t < 4; ++t) {
        for (int l = 0; l < 4; ++l) {
            const bf16* xt = (l == 0) ? (seqpad + (size_t)t * 115200)
                                      : (hpad + (size_t)(l - 1) * 115200);
            // conv_x: 8 -> 56 (pad 64)
            conv_gemm_wmma<4><<<dim3(256), 32, 0, stream>>>(
                xt, wxp, xcc, 7200L, 1, 30, 30, 1, 16, 16, 1, 15, 15, 1800, 57, 64, 0);
            layernorm_k<<<dim3(16), 256, 0, stream>>>(xcc, gx, bx, xccn, 56, 64);
            // conv_h: 8 -> 32
            conv_gemm_wmma<2><<<dim3(256), 32, 0, stream>>>(
                hpad + (size_t)l * 115200, whp, hcc, 7200L,
                1, 30, 30, 1, 16, 16, 1, 15, 15, 1800, 57, 32, 0);
            layernorm_k<<<dim3(16), 256, 0, stream>>>(hcc, gh, bh, hccn, 32, 32);
            // conv_m: 8 -> 24 (pad 32)
            conv_gemm_wmma<2><<<dim3(256), 32, 0, stream>>>(
                mpad, wmp, mcc, 7200L, 1, 30, 30, 1, 16, 16, 1, 15, 15, 1800, 57, 32, 0);
            layernorm_k<<<dim3(16), 256, 0, stream>>>(mcc, gm, bm, mccn, 24, 32);
            // gates -> c,m (fp32) + mem (fp32) + padded bf16 copies
            gates_k<<<dim3(128), 256, 0, stream>>>(xccn, hccn, mccn,
                                                   cs + (size_t)l * 32768, mbuf, memf,
                                                   mpad, mempad);
            // conv_o: 16 -> 8 (pad 16)
            conv_gemm_wmma<1><<<dim3(256), 32, 0, stream>>>(
                mempad, wop, occ, 14400L, 1, 30, 30, 1, 16, 16, 1, 15, 15, 3600, 113, 16, 0);
            layernorm_k<<<dim3(16), 256, 0, stream>>>(occ, go, bo, occn, 8, 16);
            // h_new (includes 1x1 wlast conv); writes fp32 h + padded bf16 h
            hout_k<<<dim3(128), 256, 0, stream>>>(xccn, hccn, occn, memf, wlast,
                                                  hs + (size_t)l * 32768,
                                                  hpad + (size_t)l * 115200);
        }
    }

    // ---- head: BN + avgpool + FC + sigmoid + BCE --------------------------
    bn_pool_k<<<dim3(8), 256, 0, stream>>>(hs + 3 * 32768, bng, bnb, pooled);
    fc_loss_k<<<dim3(1), 256, 0, stream>>>(pooled, fcw, fcb, target, outp);
}
